// KBRDModel_72550587564264
// MI455X (gfx1250) — matrix operand order repaired and checked by reading, not compiled, not run on hardware
//
#include <hip/hip_runtime.h>
#include <hip/hip_bf16.h>

#define DV   128
#define LSEQ 200
#define LPAD 208     // 13 * 16
#define NV   100000
#define NB   2048

typedef __attribute__((ext_vector_type(16))) __bf16 v16bf;
typedef __attribute__((ext_vector_type(8)))  float  v8f;

union FragB16 { uint4 q[2]; unsigned int u[8]; v16bf v; };

__device__ __forceinline__ unsigned short f2bf(float f) {
  unsigned int u = __float_as_uint(f);
  u = (u + 0x7FFFu + ((u >> 16) & 1u)) >> 16;   // round-to-nearest-even
  return (unsigned short)u;
}
__device__ __forceinline__ float bf2f(unsigned short h) {
  return __uint_as_float(((unsigned int)h) << 16);
}
__device__ __forceinline__ v8f wmma_bf16(v16bf a, v16bf b, v8f c) {
  // (neg_a, A, neg_b, B, c_mod, C, reuse_a, reuse_b)
  return __builtin_amdgcn_wmma_f32_16x16x32_bf16(false, a, false, b, (short)0, c,
                                                 false, false);
}

// Branchless tanh: hardware v_tanh_f32 on gfx1250 if the builtin exists,
// else clamped (e^2x-1)/(e^2x+1) on the native exp unit (no divergence,
// no scratch spills — the OCML tanhf path was spilling in the hot loop).
__device__ __forceinline__ float fast_tanh(float x) {
#if __has_builtin(__builtin_amdgcn_tanhf)
  return __builtin_amdgcn_tanhf(x);
#else
  float xc = fminf(fmaxf(x, -9.f), 9.f);       // e^18 finite in fp32
  float t = __expf(2.f * xc);
  return (t - 1.f) * __frcp_rn(t + 1.f);
#endif
}

// -------- token_emb fp32 -> bf16 (once per launch; 25.6MB, L2 resident) -----
__global__ __launch_bounds__(256) void conv_bf16_kernel(
    const float4* __restrict__ in4, uint2* __restrict__ out2, int n4) {
  int i = blockIdx.x * 256 + threadIdx.x;
  if (i >= n4) return;
  float4 v = in4[i];
  uint2 o;
  o.x = (unsigned)f2bf(v.x) | ((unsigned)f2bf(v.y) << 16);
  o.y = (unsigned)f2bf(v.z) | ((unsigned)f2bf(v.w) << 16);
  out2[i] = o;
}

// -------- per-row attention encoder: gather + tanh(h@A)@b + softmax pool ----
__global__ __launch_bounds__(128) void encode_kernel(
    const int* __restrict__ ids,
    const int* __restrict__ lengths,
    const float* __restrict__ token_emb,
    const float* __restrict__ attn_a,
    const float* __restrict__ attn_b,
    unsigned short* __restrict__ user16) {
  extern __shared__ char smem[];
  unsigned short* At = (unsigned short*)smem;      // [DV][DV] At[n*DV+k] = A[k][n]
  unsigned short* Hs = At + DV * DV;               // [LPAD][DV] gathered embeddings
  float* e_s = (float*)(Hs + (size_t)LPAD * DV);   // [LPAD] attention logits/weights
  float* red = e_s + LPAD;                         // [128] reduction scratch

  const int b   = blockIdx.x;
  const int tid = threadIdx.x;
  const int len = lengths[b];

  // attn_a transposed into LDS (so WMMA B-fragments are K-contiguous b128 reads)
  for (int i = tid; i < DV * DV; i += 128) {
    int k = i >> 7, n = i & (DV - 1);
    At[n * DV + k] = f2bf(attn_a[i]);
  }
  // ragged gather -> bf16 LDS; pad rows use token 0 (zero embedding, masked later)
  for (int i = tid; i < LPAD * DV; i += 128) {
    int l = i >> 7, d = i & (DV - 1);
    int id = (l < len) ? ids[b * LSEQ + l] : 0;
    Hs[l * DV + d] = f2bf(token_emb[id * DV + d]);
  }
  __syncthreads();

  const int wave = tid >> 5, lane = tid & 31;
  const int ln = lane & 15, half = lane >> 4;

  // e[l] = tanh(h[l]@A) . b   via  13 L-tiles of 16, each [16x128]x[128x128]
  for (int t = wave; t < LPAD / 16; t += 4) {
    const uint4* hq = (const uint4*)(Hs + (size_t)(t * 16 + ln) * DV);
    float epart[8];
#pragma unroll
    for (int r = 0; r < 8; ++r) epart[r] = 0.f;
#pragma unroll
    for (int j = 0; j < 8; ++j) {                  // N-tiles of 16
      const uint4* bq = (const uint4*)(At + (size_t)(j * 16 + ln) * DV);
      v8f acc = {};
#pragma unroll
      for (int kk = 0; kk < 4; ++kk) {             // K = 4 x 32
        FragB16 a, bb;
        a.q[0]  = hq[4 * kk + half];               // A: K {0..7 | 8..15}+32kk
        a.q[1]  = hq[4 * kk + 2 + half];           //    K {16..23 | 24..31}+32kk
        bb.q[0] = bq[4 * kk + half];               // B: K {0..7 | 8..15}+32kk
        bb.q[1] = bq[4 * kk + 2 + half];           //    K {16..23 | 24..31}+32kk
        acc = wmma_bf16(a.v, bb.v, acc);
      }
      float bn = attn_b[j * 16 + ln];
#pragma unroll
      for (int r = 0; r < 8; ++r) epart[r] += fast_tanh(acc[r]) * bn;
    }
#pragma unroll
    for (int m = 8; m >= 1; m >>= 1)
#pragma unroll
      for (int r = 0; r < 8; ++r) epart[r] += __shfl_xor(epart[r], m, 32);
    if (ln == 0) {
      int base = t * 16 + half * 8;                // C rows: VGPR r -> M = 8*half + r
#pragma unroll
      for (int r = 0; r < 8; ++r) e_s[base + r] = epart[r];
    }
  }
  __syncthreads();

  // masked softmax over e_s[0..len)
  float lmax = -1e30f;
  for (int l = tid; l < LPAD; l += 128)
    lmax = fmaxf(lmax, (l < len) ? e_s[l] : -1e30f);
  red[tid] = lmax; __syncthreads();
  for (int s = 64; s > 0; s >>= 1) {
    if (tid < s) red[tid] = fmaxf(red[tid], red[tid + s]);
    __syncthreads();
  }
  const float mx = red[0];
  __syncthreads();
  float lsum = 0.f;
  for (int l = tid; l < LPAD; l += 128) {
    float p = (l < len) ? __expf(e_s[l] - mx) : 0.f;
    e_s[l] = p; lsum += p;
  }
  red[tid] = lsum; __syncthreads();
  for (int s = 64; s > 0; s >>= 1) {
    if (tid < s) red[tid] += red[tid + s];
    __syncthreads();
  }
  const float inv = 1.f / red[0];

  // user[b][d] = sum_l attn[l] * h[l][d]   (thread = dim)
  float acc = 0.f;
  for (int l = 0; l < len; ++l) acc += e_s[l] * bf2f(Hs[l * DV + tid]);
  user16[b * DV + tid] = f2bf(acc * inv);
}

// -------- logits = user @ emb^T + bias, fused exp-sum partials --------------
__global__ __launch_bounds__(256) void logits_kernel(
    const unsigned short* __restrict__ user16,
    const unsigned short* __restrict__ emb16,
    const float* __restrict__ rec_bias,
    float* __restrict__ logits, float* __restrict__ rowsum) {
  const int tid = threadIdx.x;
  const int w = tid >> 5, lane = tid & 31, ln = lane & 15, half = lane >> 4;
  const int  mrow0 = blockIdx.y * 32 + (w >> 2) * 16;      // wave's 16-row tile
  const long nb0   = (long)blockIdx.x * 512 + (w & 3) * 128;

  // A-fragments: this wave's 16 user rows, all 4 K-steps (loaded once)
  const uint4* aq = (const uint4*)(user16 + (size_t)(mrow0 + ln) * DV);
  FragB16 af[4];
#pragma unroll
  for (int kk = 0; kk < 4; ++kk) {
    af[kk].q[0] = aq[4 * kk + half];
    af[kk].q[1] = aq[4 * kk + 2 + half];
  }
  float sump[8];
#pragma unroll
  for (int r = 0; r < 8; ++r) sump[r] = 0.f;

#pragma unroll 1
  for (int jj = 0; jj < 8; ++jj) {                         // 8 N-subtiles of 16
    long n = nb0 + jj * 16 + ln;
    bool valid = (n < NV);
    long nc = valid ? n : (NV - 1);                        // clamp: EXEC stays full
    const uint4* bq = (const uint4*)(emb16 + nc * DV);     // emb row == B column
    v8f acc = {};
#pragma unroll
    for (int kk = 0; kk < 4; ++kk) {
      FragB16 bf;
      bf.q[0] = bq[4 * kk + half];
      bf.q[1] = bq[4 * kk + 2 + half];
      acc = wmma_bf16(af[kk].v, bf.v, acc);
    }
    float bias = rec_bias[nc];
#pragma unroll
    for (int r = 0; r < 8; ++r) {
      float lg = acc[r] + bias;
      if (valid) {
        logits[(long)(mrow0 + half * 8 + r) * NV + n] = lg;
        sump[r] += __expf(lg);                             // logits are O(1): safe
      }
    }
  }
  // per-row exp-sum: reduce across the 16 lanes of each half, one atomic per row
#pragma unroll
  for (int m = 8; m >= 1; m >>= 1)
#pragma unroll
    for (int r = 0; r < 8; ++r) sump[r] += __shfl_xor(sump[r], m, 32);
  if (ln == 0) {
#pragma unroll
    for (int r = 0; r < 8; ++r)
      atomicAdd(&rowsum[mrow0 + half * 8 + r], sump[r]);
  }
}

// -------- final NLL + labels-as-float output --------------------------------
__global__ __launch_bounds__(1024) void loss_kernel(
    const int* __restrict__ labels, const float* __restrict__ logits,
    const float* __restrict__ rowsum, float* __restrict__ out_tail) {
  __shared__ float red[1024];
  const int tid = threadIdx.x;
  float part = 0.f;
  for (int b = tid; b < NB; b += 1024) {
    int lbl = labels[b];
    part += logits[(long)b * NV + lbl] - __logf(rowsum[b]);
    out_tail[b] = (float)lbl;
  }
  red[tid] = part; __syncthreads();
  for (int s = 512; s > 0; s >>= 1) {
    if (tid < s) red[tid] += red[tid + s];
    __syncthreads();
  }
  if (tid == 0) out_tail[NB] = -red[0] / (float)NB;
}

extern "C" void kernel_launch(void* const* d_in, const int* in_sizes, int n_in,
                              void* d_out, int out_size, void* d_ws, size_t ws_size,
                              hipStream_t stream) {
  const int*   ids      = (const int*)d_in[0];
  const int*   lengths  = (const int*)d_in[1];
  const int*   labels   = (const int*)d_in[2];
  const float* tokemb   = (const float*)d_in[3];
  const float* attn_a   = (const float*)d_in[4];
  const float* attn_b   = (const float*)d_in[5];
  const float* rec_bias = (const float*)d_in[6];
  float* out = (float*)d_out;

  char* ws = (char*)d_ws;
  unsigned short* emb16  = (unsigned short*)ws;                          // V*D*2 = 25.6MB
  unsigned short* user16 = (unsigned short*)(ws + (size_t)NV * DV * 2);  // B*D*2
  float* rowsum = (float*)(ws + (size_t)NV * DV * 2 + (size_t)NB * DV * 2);

  hipMemsetAsync(rowsum, 0, NB * sizeof(float), stream);

  int n4 = NV * DV / 4;
  conv_bf16_kernel<<<(n4 + 255) / 256, 256, 0, stream>>>(
      (const float4*)tokemb, (uint2*)emb16, n4);

  size_t smem = (size_t)(DV * DV + LPAD * DV) * 2 + (size_t)(LPAD + 128) * 4; // 87360B
  encode_kernel<<<NB, 128, smem, stream>>>(ids, lengths, tokemb, attn_a, attn_b,
                                           user16);

  dim3 g((NV + 511) / 512, NB / 32);   // 196 x 64
  logits_kernel<<<g, 256, 0, stream>>>(user16, emb16, rec_bias, out, rowsum);

  loss_kernel<<<1, 1024, 0, stream>>>(labels, out, rowsum,
                                      out + (size_t)NB * NV);
}
